// DRMM_71829033058371
// MI455X (gfx1250) — matrix operand order repaired and checked by reading, not compiled, not run on hardware
//
#include <hip/hip_runtime.h>
#include <hip/hip_bf16.h>
#include <math.h>

// ---------------- problem constants (match reference) ----------------
constexpr int B_  = 32;
constexpr int D_  = 16;
constexpr int L_  = 512;
constexpr int Q_  = 8;
constexpr int E_  = 128;
constexpr int NB_ = 30;
constexpr float kBnInv  = 0.99950037f;          // 1/sqrt(1 + 1e-3)
constexpr float kLo     = 0.001f;
constexpr float kBinScl = 30.0f / 0.999f;       // NB / (HI - LO)

typedef __attribute__((ext_vector_type(16))) __bf16 v16bf;
typedef __attribute__((ext_vector_type(8)))  float  v8f;

// ws layout: [gate 256 f32][q_norm 256 f32][B-panel 32*4*32*16 bf16]
#define WS_PANEL_OFF 512  // floats before panel

// =====================================================================
// Kernel A: gate softmax, query norms, and bf16 WMMA B-panel packing.
// One block, 256 threads (t maps exactly to (b,q) for the gate part).
// Panel fragment order: panel[b][ks][lane][i], K=ks*32+(i>>3)*16+(lane>>4)*8+(i&7),
// col = lane&15 (cols 8..15 are zero pad).
// =====================================================================
__global__ __launch_bounds__(256)
void drmm_prep_kernel(const int* __restrict__ query,
                      const float* __restrict__ emb,
                      const float* __restrict__ idf,
                      const float* __restrict__ w_gate,
                      float* __restrict__ ws)
{
    __shared__ float sm[256];
    const int t = threadIdx.x;
    const int b = t >> 3;
    const int q = t & 7;

    // ---- gate = softmax_q(w_gate[q] * idf[query[b,q]]) ----
    const int qidx = query[t];                 // [B,Q] flat, 256 entries
    const float s  = w_gate[q] * idf[qidx];
    sm[t] = s;
    __syncthreads();
    float mx = -1e30f;
    #pragma unroll
    for (int j = 0; j < Q_; ++j) mx = fmaxf(mx, sm[(b << 3) + j]);
    float sum = 0.0f;
    #pragma unroll
    for (int j = 0; j < Q_; ++j) sum += __expf(sm[(b << 3) + j] - mx);
    ws[t] = __expf(s - mx) * __builtin_amdgcn_rcpf(sum);  // gate[b*8+q]

    // ---- q_norm[b,q] ----
    const float* qrow = emb + (size_t)qidx * E_;
    float n2 = 0.0f;
    #pragma unroll 4
    for (int e = 0; e < E_; ++e) { float v = qrow[e]; n2 += v * v; }
    ws[256 + t] = __builtin_amdgcn_sqrtf(n2);

    // ---- bf16 B-panel in WMMA fragment order ----
    __bf16* panel = (__bf16*)(ws + WS_PANEL_OFF);
    for (int idx = t; idx < B_ * 4 * 32 * 16; idx += 256) {
        const int i    = idx & 15;
        const int lane = (idx >> 4) & 31;
        const int ks   = (idx >> 9) & 3;
        const int bb   = idx >> 11;
        const int col  = lane & 15;
        const int K    = ks * 32 + ((i >> 3) << 4) + ((lane >> 4) << 3) + (i & 7);
        float v = 0.0f;
        if (col < Q_) {
            const int qi = query[bb * Q_ + col];
            v = emb[(size_t)qi * E_ + K];
        }
        panel[idx] = (__bf16)v;
    }
}

// =====================================================================
// Kernel B: fused gather -> bf16 WMMA dots -> cosine -> histogram ->
// BN/MLP (30->128->64->32->1, tanh) -> gated sum.  One block per (b,d):
// 512 blocks x 256 threads (8 waves, 4 M-tiles of 16 tokens per wave).
// =====================================================================
__global__ __launch_bounds__(256)
void drmm_main_kernel(const int*   __restrict__ doc,
                      const float* __restrict__ emb,
                      const float* __restrict__ ws,
                      const float* __restrict__ g_in, const float* __restrict__ b_in,
                      const float* __restrict__ W0, const float* __restrict__ b0,
                      const float* __restrict__ g0, const float* __restrict__ be0,
                      const float* __restrict__ W1, const float* __restrict__ b1,
                      const float* __restrict__ g1, const float* __restrict__ be1,
                      const float* __restrict__ W2, const float* __restrict__ b2,
                      const float* __restrict__ g2, const float* __restrict__ be2,
                      const float* __restrict__ W3, const float* __restrict__ b3,
                      const float* __restrict__ g3, const float* __restrict__ be3,
                      float* __restrict__ out)
{
    __shared__ __align__(32) __bf16 bpanel[4 * 32 * 16];  // 4 KB
    __shared__ unsigned hist[Q_ * 32];                    // stride 32
    __shared__ float qn_s[Q_], gate_s[Q_];
    __shared__ float xin[Q_ * NB_];
    __shared__ float act0[Q_ * 128];
    __shared__ float act1[Q_ * 64];
    __shared__ float act2[Q_ * 32];
    __shared__ float accq[Q_];

    const int t    = threadIdx.x;
    const int bid  = blockIdx.x;          // bid = b*D + d
    const int b    = bid >> 4;
    const int wave = t >> 5;
    const int lane = t & 31;

    // stage B-panel for this batch, gate/qnorm, zero histogram
    {
        const unsigned* src = (const unsigned*)(ws + WS_PANEL_OFF) + b * 1024;
        unsigned* dst = (unsigned*)bpanel;
        #pragma unroll
        for (int r = 0; r < 4; ++r) dst[t + 256 * r] = src[t + 256 * r];
        hist[t] = 0u;
        if (t < Q_) { gate_s[t] = ws[b * Q_ + t]; qn_s[t] = ws[256 + b * Q_ + t]; }
    }
    __syncthreads();

    // ---------------- phase 1: WMMA dots + histogram ----------------
    const int* docrow = doc + (size_t)bid * L_;
    const int col  = lane & 15;
    const int mhi  = (lane >> 4) << 3;         // C rows this lane holds: mhi+j
    const int koff = (lane >> 4) << 3;         // A-frag K sub-offset (0 or 8)
    const float qn = (col < Q_) ? qn_s[col] : 1.0f;

    for (int mt = wave; mt < L_ / 16; mt += 8) {
        const int m0  = mt * 16;
        const int tok = docrow[m0 + (lane & 15)];
        const float* erow = emb + (size_t)tok * E_;

        v8f acc = {};
        float dn2 = 0.0f;
        #pragma unroll
        for (int ks = 0; ks < 4; ++ks) {
            const int K0 = ks * 32 + koff;
            const float4 a0 = *(const float4*)(erow + K0);
            const float4 a1 = *(const float4*)(erow + K0 + 4);
            const float4 a2 = *(const float4*)(erow + K0 + 16);
            const float4 a3 = *(const float4*)(erow + K0 + 20);
            dn2 += a0.x*a0.x + a0.y*a0.y + a0.z*a0.z + a0.w*a0.w
                 + a1.x*a1.x + a1.y*a1.y + a1.z*a1.z + a1.w*a1.w
                 + a2.x*a2.x + a2.y*a2.y + a2.z*a2.z + a2.w*a2.w
                 + a3.x*a3.x + a3.y*a3.y + a3.z*a3.z + a3.w*a3.w;
            v16bf af;
            af[0]=(__bf16)a0.x;  af[1]=(__bf16)a0.y;  af[2]=(__bf16)a0.z;  af[3]=(__bf16)a0.w;
            af[4]=(__bf16)a1.x;  af[5]=(__bf16)a1.y;  af[6]=(__bf16)a1.z;  af[7]=(__bf16)a1.w;
            af[8]=(__bf16)a2.x;  af[9]=(__bf16)a2.y;  af[10]=(__bf16)a2.z; af[11]=(__bf16)a2.w;
            af[12]=(__bf16)a3.x; af[13]=(__bf16)a3.y; af[14]=(__bf16)a3.z; af[15]=(__bf16)a3.w;
            const v16bf bf = *(const v16bf*)(bpanel + (ks * 32 + lane) * 16);
            acc = __builtin_amdgcn_wmma_f32_16x16x32_bf16(
                      false, af, false, bf, (short)0, acc, false, false);
        }
        // full ||d||^2: lanes L and L+16 each hold half the K range
        dn2 += __shfl_xor(dn2, 16, 32);
        const float dn = __builtin_amdgcn_sqrtf(dn2);   // raw v_sqrt_f32

        #pragma unroll
        for (int j = 0; j < 8; ++j) {
            const float dm = __shfl(dn, mhi + j, 32);   // norm of row mhi+j
            if (col < Q_) {
                // v_rcp_f32 instead of IEEE divide: bin only needs ~1e-3 res
                const float sim = acc[j] * __builtin_amdgcn_rcpf(dm * qn + 1e-8f);
                int bin = (int)floorf((sim - kLo) * kBinScl);
                bin = min(max(bin, 0), NB_ - 1);
                atomicAdd(&hist[col * 32 + bin], 1u);
            }
        }
    }
    __syncthreads();

    // ---------------- phase 2: BN + 4-layer MLP per query row ----------------
    if (t < Q_ * NB_) {
        const int q = t / NB_, k = t - q * NB_;
        xin[t] = g_in[k] * ((float)hist[q * 32 + k] * kBnInv) + b_in[k];
    }
    __syncthreads();

    for (int o = t; o < Q_ * 128; o += 256) {            // 30 -> 128
        const int q = o >> 7, j = o & 127;
        float s = b0[j];
        #pragma unroll 5
        for (int k = 0; k < NB_; ++k) s += xin[q * NB_ + k] * W0[k * 128 + j];
        act0[o] = tanhf(g0[j] * (s * kBnInv) + be0[j]);
    }
    __syncthreads();

    for (int o = t; o < Q_ * 64; o += 256) {             // 128 -> 64
        const int q = o >> 6, j = o & 63;
        float s = b1[j];
        #pragma unroll 8
        for (int k = 0; k < 128; ++k) s += act0[q * 128 + k] * W1[k * 64 + j];
        act1[o] = tanhf(g1[j] * (s * kBnInv) + be1[j]);
    }
    __syncthreads();

    if (t < Q_ * 32) {                                   // 64 -> 32
        const int q = t >> 5, j = t & 31;
        float s = b2[j];
        #pragma unroll 8
        for (int k = 0; k < 64; ++k) s += act1[q * 64 + k] * W2[k * 32 + j];
        act2[t] = tanhf(g2[j] * (s * kBnInv) + be2[j]);
    }
    __syncthreads();

    if (t < Q_) {                                        // 32 -> 1, gate
        float s = b3[0];
        #pragma unroll 8
        for (int k = 0; k < 32; ++k) s += act2[t * 32 + k] * W3[k];
        const float y = tanhf(g3[0] * (s * kBnInv) + be3[0]);
        accq[t] = y * gate_s[t];
    }
    __syncthreads();

    if (t == 0) {
        float tot = 0.0f;
        #pragma unroll
        for (int q = 0; q < Q_; ++q) tot += accq[q];
        out[bid] = tot;
    }
}

// =====================================================================
extern "C" void kernel_launch(void* const* d_in, const int* in_sizes, int n_in,
                              void* d_out, int out_size, void* d_ws, size_t ws_size,
                              hipStream_t stream) {
    const int*   doc   = (const int*)  d_in[0];
    const int*   query = (const int*)  d_in[1];
    const float* emb   = (const float*)d_in[2];
    const float* idf   = (const float*)d_in[3];
    const float* wg    = (const float*)d_in[4];
    const float* g_in  = (const float*)d_in[5];
    const float* b_in  = (const float*)d_in[6];
    const float* W0 = (const float*)d_in[7],  *b0 = (const float*)d_in[8];
    const float* g0 = (const float*)d_in[9],  *be0 = (const float*)d_in[10];
    const float* W1 = (const float*)d_in[11], *b1 = (const float*)d_in[12];
    const float* g1 = (const float*)d_in[13], *be1 = (const float*)d_in[14];
    const float* W2 = (const float*)d_in[15], *b2 = (const float*)d_in[16];
    const float* g2 = (const float*)d_in[17], *be2 = (const float*)d_in[18];
    const float* W3 = (const float*)d_in[19], *b3 = (const float*)d_in[20];
    const float* g3 = (const float*)d_in[21], *be3 = (const float*)d_in[22];
    float* out = (float*)d_out;
    float* ws  = (float*)d_ws;

    drmm_prep_kernel<<<1, 256, 0, stream>>>(query, emb, idf, wg, ws);
    drmm_main_kernel<<<B_ * D_, 256, 0, stream>>>(
        doc, emb, ws, g_in, b_in,
        W0, b0, g0, be0, W1, b1, g1, be1,
        W2, b2, g2, be2, W3, b3, g3, be3, out);
}
